// MoE_24215025615347
// MI455X (gfx1250) — matrix operand order repaired and checked by reading, not compiled, not run on hardware
//
#include <hip/hip_runtime.h>

typedef unsigned short u16;
typedef __attribute__((ext_vector_type(16))) __bf16 v16bf;
typedef __attribute__((ext_vector_type(8)))  __bf16 v8bf;
typedef __attribute__((ext_vector_type(8)))  float   v8f;

#define H_DIM 1024
#define E_NUM 8
#define T_TOK 8192
#define BM 128
#define BN 128
#define BK 32
#define NK (H_DIM / BK)
#define LDP 40   // padded LDS row stride (elements): 80B, 16B-aligned

union Frag { v16bf v; v8bf h[2]; };

__device__ __forceinline__ u16 f2bf(float f) {
    union { float f; unsigned u; } v; v.f = f;
    unsigned u = v.u;
    u += 0x7FFFu + ((u >> 16) & 1u);   // round-to-nearest-even
    return (u16)(u >> 16);
}
__device__ __forceinline__ float bf2f(u16 b) {
    union { unsigned u; float f; } v; v.u = ((unsigned)b) << 16;
    return v.f;
}
__device__ __forceinline__ unsigned lds_addr(const void* p) {
    // generic pointers to __shared__ carry the LDS byte offset in the low 32 bits
    return (unsigned)(unsigned long long)p;
}
// Copy one 64B row (global -> LDS) with a single address pair; INST_OFFSET is
// applied to BOTH the LDS and global addresses per the async pseudocode.
__device__ __forceinline__ void async_copy_row64(unsigned lds_dst, const void* gsrc) {
    asm volatile("global_load_async_to_lds_b128 %0, %1, off\n\t"
                 "global_load_async_to_lds_b128 %0, %1, off offset:16\n\t"
                 "global_load_async_to_lds_b128 %0, %1, off offset:32\n\t"
                 "global_load_async_to_lds_b128 %0, %1, off offset:48"
                 :: "v"(lds_dst), "v"(gsrc) : "memory");
}
__device__ __forceinline__ void wait_async0() {
    asm volatile("s_wait_asynccnt 0x0" ::: "memory");
}

// ---------------- 1: zero output + counters ----------------
__global__ void zero_kernel(float* out, size_t n, int* counts, float* meanp) {
    size_t i = (size_t)blockIdx.x * blockDim.x + threadIdx.x;
    if (i < n) out[i] = 0.0f;
    if (i < E_NUM) { counts[i] = 0; meanp[i] = 0.0f; }
}

// ---------------- 2: weights fp32 -> bf16, TRANSPOSED ----------------
__global__ void cvt_wt_kernel(const float* __restrict__ w1, const float* __restrict__ We,
                              u16* __restrict__ w1t, u16* __restrict__ wet) {
    __shared__ u16 tile[32][33];
    const int mtx = blockIdx.z;
    const float* src = (mtx == 0) ? w1 : We + (size_t)(mtx - 1) * H_DIM * H_DIM;
    u16* dst = (mtx == 0) ? w1t : wet + (size_t)(mtx - 1) * H_DIM * H_DIM;
    const int k0 = blockIdx.y * 32, n0 = blockIdx.x * 32;
    const int tx = threadIdx.x & 31, ty = threadIdx.x >> 5;
#pragma unroll
    for (int i = 0; i < 4; i++) {
        int k = ty + i * 8;
        tile[k][tx] = f2bf(src[(size_t)(k0 + k) * H_DIM + n0 + tx]);
    }
    __syncthreads();
#pragma unroll
    for (int i = 0; i < 4; i++) {
        int n = ty + i * 8;
        dst[(size_t)(n0 + n) * H_DIM + k0 + tx] = tile[tx][n];
    }
}

// ---------------- 3: LayerNorm + bf16 convert ----------------
__global__ void ln_cvt_kernel(const float* __restrict__ x, const float* __restrict__ scale,
                              const float* __restrict__ bias,
                              u16* __restrict__ xbf, u16* __restrict__ xnbf) {
    int t = blockIdx.x, tid = threadIdx.x;
    const float* row = x + (size_t)t * H_DIM;
    float v[4], s = 0.f, s2 = 0.f;
#pragma unroll
    for (int i = 0; i < 4; i++) {
        v[i] = row[tid + 256 * i];
        s += v[i]; s2 += v[i] * v[i];
    }
    __shared__ float rs[256], rq[256];
    rs[tid] = s; rq[tid] = s2; __syncthreads();
    for (int st = 128; st > 0; st >>= 1) {
        if (tid < st) { rs[tid] += rs[tid + st]; rq[tid] += rq[tid + st]; }
        __syncthreads();
    }
    float mean = rs[0] * (1.0f / H_DIM);
    float var  = rq[0] * (1.0f / H_DIM) - mean * mean;
    float rstd = rsqrtf(var + 1e-5f);
#pragma unroll
    for (int i = 0; i < 4; i++) {
        int h = tid + 256 * i;
        float xn = (v[i] - mean) * rstd * scale[h] + bias[h];
        xnbf[(size_t)t * H_DIM + h] = f2bf(xn);
        xbf [(size_t)t * H_DIM + h] = f2bf(v[i]);
    }
}

// ---------------- 4: router GEMM  hmid = relu(xn @ w1 + b1), bf16 out ----------------
// 128 threads = 4 waves (2x2), each wave computes 64x64 (4x4 WMMA tiles).
__global__ __launch_bounds__(128) void gemm_router_kernel(
        const u16* __restrict__ xnbf, const u16* __restrict__ w1t,
        const float* __restrict__ b1, u16* __restrict__ hmid) {
    __shared__ __attribute__((aligned(16))) __bf16 Als[2][BM][LDP];
    __shared__ __attribute__((aligned(16))) __bf16 Bls[2][BN][LDP];
    const int tid = threadIdx.x, lane = tid & 31, wave = tid >> 5;
    const int wm = wave >> 1, wn = wave & 1;     // 2 x 2 wave grid
    const size_t mbase = (size_t)blockIdx.x * BM;
    const int nbase = blockIdx.y * BN;
    const u16* gAp = xnbf + (mbase + tid) * H_DIM;            // row tid of A tile
    const u16* gBp = w1t + (size_t)(nbase + tid) * H_DIM;     // row tid of B tile
    const unsigned ldsA[2] = { lds_addr(&Als[0][tid][0]), lds_addr(&Als[1][tid][0]) };
    const unsigned ldsB[2] = { lds_addr(&Bls[0][tid][0]), lds_addr(&Bls[1][tid][0]) };

    v8f acc[4][4] = {};
    async_copy_row64(ldsA[0], gAp);       // stage k-tile 0
    async_copy_row64(ldsB[0], gBp);
    const int k0 = (lane >> 4) * 8, kb = (lane >> 4) * 16, l15 = lane & 15;
    for (int kt = 0; kt < NK; kt++) {
        const int cur = kt & 1;
        wait_async0();          // my async writes into buf[cur] have landed
        __syncthreads();        // everyone's have; prior readers of buf[cur^1] done
        if (kt + 1 < NK) {
            async_copy_row64(ldsA[cur ^ 1], gAp + BK);
            async_copy_row64(ldsB[cur ^ 1], gBp + BK);
        }
        if (kt + 2 < NK) {
            __builtin_prefetch(gAp + 2 * BK, 0, 1);
            __builtin_prefetch(gBp + 2 * BK, 0, 1);
        }
        Frag a[4], b[4];
#pragma unroll
        for (int mt = 0; mt < 4; mt++) {
            int m = wm * 64 + mt * 16 + l15;
            a[mt].h[0] = *(const v8bf*)&Als[cur][m][k0];
            a[mt].h[1] = *(const v8bf*)&Als[cur][m][k0 + 16];
        }
#pragma unroll
        for (int nt = 0; nt < 4; nt++) {
            int n = wn * 64 + nt * 16 + l15;
            b[nt].h[0] = *(const v8bf*)&Bls[cur][n][k0];
            b[nt].h[1] = *(const v8bf*)&Bls[cur][n][k0 + 16];
        }
#pragma unroll
        for (int mt = 0; mt < 4; mt++)
#pragma unroll
            for (int nt = 0; nt < 4; nt++)
                acc[mt][nt] = __builtin_amdgcn_wmma_f32_16x16x32_bf16(
                    false, a[mt].v, false, b[nt].v, (short)0, acc[mt][nt], false, false);
        gAp += BK; gBp += BK;
    }
    const int rbase = (lane >> 4) * 8;
#pragma unroll
    for (int mt = 0; mt < 4; mt++)
#pragma unroll
        for (int nt = 0; nt < 4; nt++) {
            int col = nbase + wn * 64 + nt * 16 + l15;
            float bb = b1[col];
            int mloc = wm * 64 + mt * 16 + rbase;
#pragma unroll
            for (int rr = 0; rr < 8; rr++) {
                float vv = acc[mt][nt][rr] + bb;
                vv = vv > 0.f ? vv : 0.f;
                hmid[(mbase + mloc + rr) * H_DIM + col] = f2bf(vv);
            }
        }
}

// ---------------- 5: logits + softmax + top2 + dispatch lists ----------------
__global__ void router_topk_kernel(const u16* __restrict__ hmid, const float* __restrict__ w2,
                                   const float* __restrict__ b2,
                                   int* counts, int* idxl, float* gatel, float* meanp) {
    int t = blockIdx.x, tid = threadIdx.x;
    float acc[E_NUM] = {};
#pragma unroll
    for (int i = 0; i < 4; i++) {
        int h = tid + 256 * i;
        float hv = bf2f(hmid[(size_t)t * H_DIM + h]);
        const float* wr = w2 + (size_t)h * E_NUM;
#pragma unroll
        for (int e = 0; e < E_NUM; e++) acc[e] += hv * wr[e];
    }
    __shared__ float red[256][E_NUM];
#pragma unroll
    for (int e = 0; e < E_NUM; e++) red[tid][e] = acc[e];
    __syncthreads();
    for (int st = 128; st > 0; st >>= 1) {
        if (tid < st)
#pragma unroll
            for (int e = 0; e < E_NUM; e++) red[tid][e] += red[tid + st][e];
        __syncthreads();
    }
    if (tid == 0) {
        float l[E_NUM], p[E_NUM], mx = -1e30f, sum = 0.f;
#pragma unroll
        for (int e = 0; e < E_NUM; e++) { l[e] = red[0][e] + b2[e]; mx = l[e] > mx ? l[e] : mx; }
#pragma unroll
        for (int e = 0; e < E_NUM; e++) { p[e] = __expf(l[e] - mx); sum += p[e]; }
        float inv = 1.0f / sum;
#pragma unroll
        for (int e = 0; e < E_NUM; e++) p[e] *= inv;
        int i0 = 0;
#pragma unroll
        for (int e = 1; e < E_NUM; e++) if (p[e] > p[i0]) i0 = e;
        int i1 = (i0 == 0) ? 1 : 0;
#pragma unroll
        for (int e = 0; e < E_NUM; e++) if (e != i0 && p[e] > p[i1]) i1 = e;
        float s01 = p[i0] + p[i1];
        float g0 = p[i0] / s01, g1 = p[i1] / s01;
        int p0 = atomicAdd(&counts[i0], 1);
        idxl[i0 * T_TOK + p0] = t; gatel[i0 * T_TOK + p0] = g0;
        int p1 = atomicAdd(&counts[i1], 1);
        idxl[i1 * T_TOK + p1] = t; gatel[i1 * T_TOK + p1] = g1;
#pragma unroll
        for (int e = 0; e < E_NUM; e++) atomicAdd(&meanp[e], p[e]);
    }
}

// ---------------- 6: gathered expert GEMM, atomic combine ----------------
__global__ __launch_bounds__(128) void gemm_expert_kernel(
        const u16* __restrict__ xbf, const u16* __restrict__ wet,
        const float* __restrict__ be, const int* __restrict__ counts,
        const int* __restrict__ idxl, const float* __restrict__ gatel,
        float* __restrict__ out) {
    const int e = blockIdx.z;
    const int cnt = counts[e];
    if ((int)blockIdx.x * BM >= cnt) return;
    __shared__ __attribute__((aligned(16))) __bf16 Als[2][BM][LDP];
    __shared__ __attribute__((aligned(16))) __bf16 Bls[2][BN][LDP];
    __shared__ int   tok[BM];
    __shared__ float gat[BM];
    const int tid = threadIdx.x, lane = tid & 31, wave = tid >> 5;
    const int wm = wave >> 1, wn = wave & 1;
    const int nbase = blockIdx.y * BN;

    {
        int pos = blockIdx.x * BM + tid;
        if (pos < cnt) { tok[tid] = idxl[e * T_TOK + pos]; gat[tid] = gatel[e * T_TOK + pos]; }
        else           { tok[tid] = -1;                    gat[tid] = 0.0f; }
    }
    __syncthreads();

    int trow = tok[tid]; trow = trow < 0 ? 0 : trow;
    const u16* gAp = xbf + (size_t)trow * H_DIM;   // gathered row tid of A tile
    const u16* gBp = wet + (size_t)e * H_DIM * H_DIM + (size_t)(nbase + tid) * H_DIM;
    const unsigned ldsA[2] = { lds_addr(&Als[0][tid][0]), lds_addr(&Als[1][tid][0]) };
    const unsigned ldsB[2] = { lds_addr(&Bls[0][tid][0]), lds_addr(&Bls[1][tid][0]) };

    v8f acc[4][4] = {};
    async_copy_row64(ldsA[0], gAp);
    async_copy_row64(ldsB[0], gBp);
    const int k0 = (lane >> 4) * 8, kb = (lane >> 4) * 16, l15 = lane & 15;
    for (int kt = 0; kt < NK; kt++) {
        const int cur = kt & 1;
        wait_async0();
        __syncthreads();
        if (kt + 1 < NK) {
            async_copy_row64(ldsA[cur ^ 1], gAp + BK);
            async_copy_row64(ldsB[cur ^ 1], gBp + BK);
        }
        if (kt + 2 < NK) {
            __builtin_prefetch(gAp + 2 * BK, 0, 1);
            __builtin_prefetch(gBp + 2 * BK, 0, 1);
        }
        Frag a[4], b[4];
#pragma unroll
        for (int mt = 0; mt < 4; mt++) {
            int m = wm * 64 + mt * 16 + l15;
            a[mt].h[0] = *(const v8bf*)&Als[cur][m][k0];
            a[mt].h[1] = *(const v8bf*)&Als[cur][m][k0 + 16];
        }
#pragma unroll
        for (int nt = 0; nt < 4; nt++) {
            int n = wn * 64 + nt * 16 + l15;
            b[nt].h[0] = *(const v8bf*)&Bls[cur][n][k0];
            b[nt].h[1] = *(const v8bf*)&Bls[cur][n][k0 + 16];
        }
#pragma unroll
        for (int mt = 0; mt < 4; mt++)
#pragma unroll
            for (int nt = 0; nt < 4; nt++)
                acc[mt][nt] = __builtin_amdgcn_wmma_f32_16x16x32_bf16(
                    false, a[mt].v, false, b[nt].v, (short)0, acc[mt][nt], false, false);
        gAp += BK; gBp += BK;
    }
    const int rbase = (lane >> 4) * 8;
#pragma unroll
    for (int mt = 0; mt < 4; mt++)
#pragma unroll
        for (int nt = 0; nt < 4; nt++) {
            int col = nbase + wn * 64 + nt * 16 + l15;
            float bb = be[(size_t)e * H_DIM + col];
            int mloc = wm * 64 + mt * 16 + rbase;
#pragma unroll
            for (int rr = 0; rr < 8; rr++) {
                int token = tok[mloc + rr];
                if (token >= 0) {
                    float g = gat[mloc + rr];
                    atomicAdd(&out[(size_t)token * H_DIM + col], g * (acc[mt][nt][rr] + bb));
                }
            }
        }
}

// ---------------- 7: aux loss ----------------
__global__ void aux_kernel(const float* meanp, float* out_aux) {
    if (threadIdx.x == 0 && blockIdx.x == 0) {
        float a = 0.f;
#pragma unroll
        for (int e = 0; e < E_NUM; e++) {
            float mp = meanp[e] / (float)T_TOK;
            a += mp * logf(mp * (float)E_NUM + 1e-9f);
        }
        *out_aux = a;
    }
}

extern "C" void kernel_launch(void* const* d_in, const int* in_sizes, int n_in,
                              void* d_out, int out_size, void* d_ws, size_t ws_size,
                              hipStream_t stream) {
    (void)in_sizes; (void)n_in; (void)out_size; (void)ws_size;
    const float* x        = (const float*)d_in[0];
    const float* ln_scale = (const float*)d_in[1];
    const float* ln_bias  = (const float*)d_in[2];
    const float* w1       = (const float*)d_in[3];
    const float* b1       = (const float*)d_in[4];
    const float* w2       = (const float*)d_in[5];
    const float* b2       = (const float*)d_in[6];
    const float* We       = (const float*)d_in[7];
    const float* be       = (const float*)d_in[8];
    float* out = (float*)d_out;

    const size_t TH = (size_t)T_TOK * H_DIM;
    char* ws = (char*)d_ws;
    size_t off = 0;
    auto take = [&](size_t bytes) { void* p = ws + off; off = (off + bytes + 255) & ~(size_t)255; return p; };
    u16*  xbf   = (u16*)take(TH * 2);
    u16*  xnbf  = (u16*)take(TH * 2);
    u16*  w1t   = (u16*)take((size_t)H_DIM * H_DIM * 2);
    u16*  wet   = (u16*)take((size_t)E_NUM * H_DIM * H_DIM * 2);
    u16*  hmid  = (u16*)take(TH * 2);
    int*  counts= (int*)take(E_NUM * sizeof(int));
    float* meanp= (float*)take(E_NUM * sizeof(float));
    int*  idxl  = (int*)take((size_t)E_NUM * T_TOK * sizeof(int));
    float* gatel= (float*)take((size_t)E_NUM * T_TOK * sizeof(float));

    {
        size_t n = TH + 1;
        zero_kernel<<<dim3((unsigned)((n + 255) / 256)), dim3(256), 0, stream>>>(out, n, counts, meanp);
    }
    cvt_wt_kernel<<<dim3(H_DIM / 32, H_DIM / 32, 1 + E_NUM), dim3(256), 0, stream>>>(w1, We, w1t, wet);
    ln_cvt_kernel<<<dim3(T_TOK), dim3(256), 0, stream>>>(x, ln_scale, ln_bias, xbf, xnbf);
    gemm_router_kernel<<<dim3(T_TOK / BM, H_DIM / BN), dim3(128), 0, stream>>>(xnbf, w1t, b1, hmid);
    router_topk_kernel<<<dim3(T_TOK), dim3(256), 0, stream>>>(hmid, w2, b2, counts, idxl, gatel, meanp);
    gemm_expert_kernel<<<dim3(T_TOK / BM, H_DIM / BN, E_NUM), dim3(128), 0, stream>>>(
        xbf, wet, be, counts, idxl, gatel, out);
    aux_kernel<<<dim3(1), dim3(32), 0, stream>>>(meanp, out + TH);
}
